// YOLOV3_29798483100450
// MI455X (gfx1250) — matrix operand order repaired and checked by reading, not compiled, not run on hardware
//
#include <hip/hip_runtime.h>
#include <hip/hip_bf16.h>
#include <cstdint>

typedef _Float16 half_t;
typedef _Float16 v16h __attribute__((ext_vector_type(16)));
typedef _Float16 v8h  __attribute__((ext_vector_type(8)));
typedef float    v8f  __attribute__((ext_vector_type(8)));
typedef unsigned int v4u __attribute__((ext_vector_type(4)));
typedef int          v4i __attribute__((ext_vector_type(4)));
typedef int          v8i __attribute__((ext_vector_type(8)));

#define BATCH      8
#define NTOT_ROWS  851760LL          // 240*(13*13 + 26*26 + 52*52)
#define CAND_MAX   16384
#define POST_NMS   100
#define NMS_TOPK   400

#if defined(__has_builtin)
#if __has_builtin(__builtin_amdgcn_tensor_load_to_lds)
#define HAVE_TDM 1
#endif
#endif

// ---------------------------------------------------------------------------
// Weight packing: OIHW f32 -> [Cout][Kpad] f16, folding BN scale. Kpad = ceil32.
// ---------------------------------------------------------------------------
__global__ void pack_weights(const float* __restrict__ w, const float* __restrict__ s,
                             half_t* __restrict__ dst, int Cout, int Ktrue, int Kpad) {
    int idx = blockIdx.x * 256 + threadIdx.x;
    if (idx >= Cout * Kpad) return;
    int co = idx / Kpad;
    int kk = idx % Kpad;
    float v = 0.f;
    if (kk < Ktrue) {
        v = w[(size_t)co * Ktrue + kk];
        if (s) v *= s[co];
    }
    dst[idx] = (half_t)v;
}

__global__ void convert_f32_to_f16(const float* __restrict__ src, half_t* __restrict__ dst, long long n) {
    long long i = (long long)blockIdx.x * 256 + threadIdx.x;
    if (i < n) dst[i] = (half_t)src[i];
}

// ---------------------------------------------------------------------------
// Implicit-GEMM convolution with WMMA f16 -> f32, templated on (KS, STRIDE).
//   M = Cout, N = B*Ho*Wo, K = Cin*KS*KS (padded to 32)
//   Block = 256 threads = 8 wave32; block tile 64(M) x 128(N);
//   wave tile 16(M) x 64(N) = 4 WMMA accumulators; double-buffered LDS.
//   A (weights) tile DMA'd by the Tensor Data Mover (wave 0 issues, waits
//   TENSORcnt); B (im2col patches) gathered per-thread, column-major in LDS.
// ---------------------------------------------------------------------------
template <int KS, int STRIDE>
__global__ void __launch_bounds__(256)
conv_wmma(const half_t* __restrict__ in, const half_t* __restrict__ wpk,
          const float* __restrict__ bias, half_t* __restrict__ out,
          int Bn, int Cin, int H, int W, int Cout, int Ho, int Wo,
          int Ktrue, int Kpad, int leaky) {
    constexpr int PAD = KS / 2;
    constexpr int K2  = KS * KS;

    __shared__ __align__(32) half_t As[2][64 * 32];    // [m][k] row-major
    __shared__ __align__(32) half_t Bs[2][128 * 32];   // [n][k] column-major (k contiguous)

    const int tid   = threadIdx.x;
    const int lane  = tid & 31;
    const int wave  = tid >> 5;         // 0..7
    const int lhalf = lane & 15;
    const int hi    = lane >> 4;        // 0 or 1

    const int mblk = blockIdx.y * 64;
    const int nblk = blockIdx.x * 128;
    const int HWo  = Ho * Wo;
    const int Ntot = Bn * HWo;

    // wave -> M subtile + 4 consecutive N subtiles
    const int m0    = (wave & 3) * 16;
    const int nbase = (wave >> 2) * 64;

    v8f acc[4] = {};

    // per-thread gather assignment for B tile: column bc, rows br0..br0+15
    const int bc  = tid & 127;
    const int br0 = (tid >> 7) * 16;
    const int n_g = nblk + bc;
    const bool nvalid = (n_g < Ntot);
    int gb = 0, goh = 0, gow = 0;
    if (nvalid) {
        gb = n_g / HWo;
        int rem = n_g % HWo;
        goh = rem / Wo;
        gow = rem % Wo;
    }
    const int ih0 = goh * STRIDE - PAD;     // top-left of receptive field
    const int iw0 = gow * STRIDE - PAD;
    const half_t* inb = in + (size_t)gb * Cin * H * W;

#if !defined(HAVE_TDM)
    // fallback per-thread A-tile assignment
    const int am  = tid >> 2;
    const int ac0 = (tid & 3) * 8;
#endif

    // ---- stage the A (weights) tile: TDM 2D tile load ----
    auto stageA = [&](int buf, int kt) {
#if defined(HAVE_TDM)
        if (wave == 0) {
            // D# group0: count=1 | lds_addr | global_addr | type=2
            uint32_t lds_addr = (uint32_t)(uintptr_t)&As[buf][0];   // low 32 bits of generic = LDS offset
            uint64_t gaddr = (uint64_t)(uintptr_t)wpk + ((uint64_t)mblk * (uint64_t)Kpad + (uint64_t)kt) * 2u;
            v4u g0 = { 1u, lds_addr, (uint32_t)gaddr,
                       (uint32_t)((gaddr >> 32) & 0x01FFFFFFu) | (2u << 30) };
            // D# group1: data_size=2B; tensor dims give free OOB zero-fill at the Cout edge
            uint32_t td0 = (uint32_t)(Kpad - kt);       // remaining K
            uint32_t td1 = (uint32_t)(Cout - mblk);     // remaining rows
            v8i g1 = { (int)(1u << 16),                                   // mask=0, data_size=1 (2B)
                       (int)((td0 & 0xFFFFu) << 16),                      // abar=0 | td0.lo
                       (int)(((td0 >> 16) & 0xFFFFu) | ((td1 & 0xFFFFu) << 16)),
                       (int)(((td1 >> 16) & 0xFFFFu) | (32u << 16)),      // tile_dim0 = 32
                       (int)(64u),                                        // tile_dim1 = 64, tile_dim2 = 0
                       (int)(uint32_t)Kpad,                               // dim0 stride lo
                       0, 0 };                                            // stride hi / dim1 stride
            v4i z4 = {};
#if defined(__clang_major__) && __clang_major__ >= 23
            v8i z8 = {};
            __builtin_amdgcn_tensor_load_to_lds(g0, g1, z4, z4, z8, 0);
#else
            __builtin_amdgcn_tensor_load_to_lds(g0, g1, z4, z4, 0);
#endif
        }
#else
        v8h av = {};
        int gco = mblk + am;
        if (gco < Cout) av = *(const v8h*)&wpk[(size_t)gco * Kpad + kt + ac0];
        *(v8h*)&As[buf][am * 32 + ac0] = av;
#endif
    };

    // ---- stage the B (im2col) tile ----
    auto stageB = [&](int buf, int kt) {
        if (KS == 1) {
            // 1x1/s1: contiguous channel run, no spatial checks needed
            const half_t* p = inb + (size_t)(kt + br0) * H * W + (size_t)goh * W + gow;
            #pragma unroll
            for (int j = 0; j < 16; ++j) {
                int ci = kt + br0 + j;
                half_t v = (half_t)0.f;
                if (nvalid && ci < Ktrue) v = p[(size_t)j * H * W];
                Bs[buf][bc * 32 + br0 + j] = v;
            }
        } else {
            #pragma unroll
            for (int j = 0; j < 16; ++j) {
                int kk = kt + br0 + j;
                half_t v = (half_t)0.f;
                if (nvalid && kk < Ktrue) {
                    int ci = kk / K2;              // constant-divisor
                    int r2 = kk % K2;
                    int kh = r2 / KS;
                    int kw = r2 % KS;
                    int ih = ih0 + kh;
                    int iw = iw0 + kw;
                    if (ih >= 0 && ih < H && iw >= 0 && iw < W)
                        v = inb[((size_t)ci * H + ih) * W + iw];
                }
                Bs[buf][bc * 32 + br0 + j] = v;
            }
        }
    };

    // prologue: fill buffer 0
    stageB(0, 0);
    stageA(0, 0);
#if defined(HAVE_TDM)
    if (wave == 0) __builtin_amdgcn_s_wait_tensorcnt(0);
#endif
    __syncthreads();

    for (int kt = 0; kt < Kpad; kt += 32) {
        const int cb = (kt >> 5) & 1;
        const bool more = (kt + 32) < Kpad;
        if (more) {                        // prefetch next tile into other buffer
            stageB(cb ^ 1, kt + 32);
            stageA(cb ^ 1, kt + 32);
        }

        // ---- fragment loads (contiguous k = 16*hi + s order, same for A and B) ----
        // A: lanes 0-15 hold M=lane, K = 0..15; lanes 16-31: K = 16..31
        const v8h* ap = (const v8h*)&As[cb][(m0 + lhalf) * 32 + hi * 16];
        v8h a_lo = ap[0];
        v8h a_hi = ap[1];   // +8 halves
        v16h afrag;
        #pragma unroll
        for (int i = 0; i < 8; ++i) { afrag[i] = a_lo[i]; afrag[i + 8] = a_hi[i]; }

        // B: lanes 0-15 hold N=lane, K=0..15; lanes 16-31 hold N=lane-16, K=16..31
        #pragma unroll
        for (int t = 0; t < 4; ++t) {
            v16h bfrag = *(const v16h*)&Bs[cb][(nbase + t * 16 + lhalf) * 32 + hi * 16];
            acc[t] = __builtin_amdgcn_wmma_f32_16x16x32_f16(false, afrag, false, bfrag,
                                                            (short)0, acc[t], false, false);
        }

#if defined(HAVE_TDM)
        if (more && wave == 0) __builtin_amdgcn_s_wait_tensorcnt(0);
#endif
        __syncthreads();
    }

    // ---- epilogue: bias + leaky ReLU, store f16 NCHW ----
    #pragma unroll
    for (int t = 0; t < 4; ++t) {
        int n = nblk + nbase + t * 16 + lhalf;
        if (n >= Ntot) continue;
        int b  = n / HWo;
        int rm = n % HWo;
        int oh = rm / Wo;
        int ow = rm % Wo;
        #pragma unroll
        for (int r = 0; r < 8; ++r) {
            int co = mblk + m0 + r + hi * 8;
            if (co >= Cout) continue;
            float v = acc[t][r] + bias[co];
            if (leaky) v = (v > 0.f) ? v : 0.1f * v;
            out[(((size_t)b * Cout + co) * Ho + oh) * Wo + ow] = (half_t)v;
        }
    }
}

// ---------------------------------------------------------------------------
// Nearest-neighbor x2 upsample + channel concat (NCHW, f16)
// ---------------------------------------------------------------------------
__global__ void upcat_kernel(const half_t* __restrict__ h, const half_t* __restrict__ route,
                             half_t* __restrict__ out, int Bn, int C1, int C2, int Hr, int Wr) {
    long long idx = (long long)blockIdx.x * 256 + threadIdx.x;
    long long tot = (long long)Bn * (C1 + C2) * Hr * Wr;
    if (idx >= tot) return;
    int x = idx % Wr;
    int y = (idx / Wr) % Hr;
    int c = (idx / ((long long)Wr * Hr)) % (C1 + C2);
    int b = idx / ((long long)Wr * Hr * (C1 + C2));
    half_t v;
    if (c < C1) v = h[(((size_t)b * C1 + c) * (Hr / 2) + (y >> 1)) * (Wr / 2) + (x >> 1)];
    else        v = route[(((size_t)b * C2 + (c - C1)) * Hr + y) * Wr + x];
    out[idx] = v;
}

// ---------------------------------------------------------------------------
// Decode: raw (B,255,H,W) f16 -> det rows (class, score, x1,y1,x2,y2) f32
// ---------------------------------------------------------------------------
__device__ inline float sigm(float x) { return 1.f / (1.f + __expf(-x)); }

__global__ void decode_kernel(const half_t* __restrict__ raw, float* __restrict__ det,
                              int Bn, int H, int W, float stride,
                              float aw0, float ah0, float aw1, float ah1, float aw2, float ah2,
                              long long baseRow, long long NtotRows) {
    long long idx = (long long)blockIdx.x * 256 + threadIdx.x;
    long long tot = (long long)Bn * H * W * 240;
    if (idx >= tot) return;
    int cls = idx % 80;
    int a   = (idx / 80) % 3;
    long long p3 = idx / 240;
    int pos = p3 % ((long long)H * W);
    int b   = p3 / ((long long)H * W);
    int oh = pos / W, ow = pos % W;
    size_t plane = (size_t)H * W;
    const half_t* rb = raw + ((size_t)b * 255 + a * 85) * plane + (size_t)oh * W + ow;
    float tx = (float)rb[0];
    float ty = (float)rb[plane];
    float tw = (float)rb[2 * plane];
    float th = (float)rb[3 * plane];
    float cf = (float)rb[4 * plane];
    float tc = (float)rb[(size_t)(5 + cls) * plane];
    float aw = (a == 0) ? aw0 : ((a == 1) ? aw1 : aw2);
    float ah = (a == 0) ? ah0 : ((a == 1) ? ah1 : ah2);
    float cx = (sigm(tx) + (float)ow) * stride;
    float cy = (sigm(ty) + (float)oh) * stride;
    float hw = __expf(tw) * aw * 0.5f;
    float hh = __expf(th) * ah * 0.5f;
    float score = sigm(cf) * sigm(tc);
    long long row = (long long)b * NtotRows + baseRow + ((long long)pos * 3 + a) * 80 + cls;
    float* d = det + row * 6;
    d[0] = (float)cls; d[1] = score;
    d[2] = cx - hw; d[3] = cy - hh; d[4] = cx + hw; d[5] = cy + hh;
}

// ---------------------------------------------------------------------------
// NMS: candidate filter (score > 0.01) then per-image greedy top-400 + IoU.
// ---------------------------------------------------------------------------
__global__ void init_cnt(int* cnt) { if (threadIdx.x < BATCH) cnt[threadIdx.x] = 0; }

__global__ void cand_kernel(const float* __restrict__ det, long long Ntot,
                            float* __restrict__ cs, int* __restrict__ ci, int* __restrict__ cnt) {
    long long idx = (long long)blockIdx.x * 256 + threadIdx.x;
    if (idx >= (long long)BATCH * Ntot) return;
    float s = det[idx * 6 + 1];
    if (s > 0.01f) {
        int b = idx / Ntot;
        int i = idx % Ntot;
        int p = atomicAdd(&cnt[b], 1);
        if (p < CAND_MAX) {
            cs[(size_t)b * CAND_MAX + p] = s;
            ci[(size_t)b * CAND_MAX + p] = i;
        }
    }
}

__global__ void __launch_bounds__(256)
nms_kernel(const float* __restrict__ det, long long Ntot,
           float* __restrict__ cs, const int* __restrict__ ci, const int* __restrict__ cnt,
           float* __restrict__ out) {
    const int b = blockIdx.x;
    const int tid = threadIdx.x;
    __shared__ float sb[NMS_TOPK * 6];
    __shared__ int   skeep[NMS_TOPK];
    __shared__ float rs[256];
    __shared__ int   ri[256];

    // init outputs to -1 (ids @0, scores @800, boxes @1600)
    for (int j = tid; j < POST_NMS; j += 256) {
        out[b * POST_NMS + j] = -1.f;
        out[BATCH * POST_NMS + b * POST_NMS + j] = -1.f;
        for (int c = 0; c < 4; ++c)
            out[2 * BATCH * POST_NMS + ((size_t)b * POST_NMS + j) * 4 + c] = -1.f;
    }

    int n = cnt[b]; if (n > CAND_MAX) n = CAND_MAX;
    float* bcs = cs + (size_t)b * CAND_MAX;
    const int* bci = ci + (size_t)b * CAND_MAX;
    __syncthreads();

    // greedy top-400 by repeated argmax (sorted descending by construction)
    for (int t = 0; t < NMS_TOPK; ++t) {
        float best = -1e30f; int bi = -1;
        for (int j = tid; j < n; j += 256) {
            float v = bcs[j];
            if (v > best) { best = v; bi = j; }
        }
        rs[tid] = best; ri[tid] = bi;
        __syncthreads();
        for (int off = 128; off > 0; off >>= 1) {
            if (tid < off && rs[tid + off] > rs[tid]) { rs[tid] = rs[tid + off]; ri[tid] = ri[tid + off]; }
            __syncthreads();
        }
        if (tid == 0) {
            if (ri[0] >= 0 && rs[0] > 0.01f) {
                int i = bci[ri[0]];
                const float* dr = det + ((size_t)b * Ntot + i) * 6;
                for (int c = 0; c < 6; ++c) sb[t * 6 + c] = dr[c];
                skeep[t] = 1;
                bcs[ri[0]] = -1e30f;
            } else {
                skeep[t] = 0;
                for (int c = 0; c < 6; ++c) sb[t * 6 + c] = 0.f;
            }
        }
        __syncthreads();
    }

    // greedy suppression
    for (int i = 0; i < NMS_TOPK; ++i) {
        __syncthreads();
        if (!skeep[i]) continue;
        float cls = sb[i * 6 + 0];
        float x1 = sb[i * 6 + 2], y1 = sb[i * 6 + 3], x2 = sb[i * 6 + 4], y2 = sb[i * 6 + 5];
        float ai = fmaxf(x2 - x1, 0.f) * fmaxf(y2 - y1, 0.f);
        for (int j = i + 1 + tid; j < NMS_TOPK; j += 256) {
            if (!skeep[j] || sb[j * 6 + 0] != cls) continue;
            float jx1 = sb[j * 6 + 2], jy1 = sb[j * 6 + 3], jx2 = sb[j * 6 + 4], jy2 = sb[j * 6 + 5];
            float aj = fmaxf(jx2 - jx1, 0.f) * fmaxf(jy2 - jy1, 0.f);
            float ix1 = fmaxf(x1, jx1), iy1 = fmaxf(y1, jy1);
            float ix2 = fminf(x2, jx2), iy2 = fminf(y2, jy2);
            float inter = fmaxf(ix2 - ix1, 0.f) * fmaxf(iy2 - iy1, 0.f);
            float iou = inter / (ai + aj - inter + 1e-12f);
            if (iou > 0.45f) skeep[j] = 0;
        }
    }
    __syncthreads();

    if (tid == 0) {
        int rank = 0;
        for (int t = 0; t < NMS_TOPK && rank < POST_NMS; ++t) {
            if (!skeep[t]) continue;
            out[b * POST_NMS + rank] = sb[t * 6 + 0];
            out[BATCH * POST_NMS + b * POST_NMS + rank] = sb[t * 6 + 1];
            for (int c = 0; c < 4; ++c)
                out[2 * BATCH * POST_NMS + ((size_t)b * POST_NMS + rank) * 4 + c] = sb[t * 6 + 2 + c];
            ++rank;
        }
    }
}

// ---------------------------------------------------------------------------
// Host orchestration
// ---------------------------------------------------------------------------
// d_in flatten order (jax sorted-key pytree):
//   params.blocks[i][j].{b,s,w}  -> 18*i + 3*j + {0,1,2}      (0..53)
//   params.outs[i].{b,w}         -> 54 + 2*i + {0,1}          (54..59)
//   params.stage[i].{b,s,w}      -> 60 + 3*i + {0,1,2}        (60..74)
//   params.trans[i].{b,s,w}      -> 75 + 3*i + {0,1,2}        (75..80)
//   x                            -> 81
struct ConvL {
    const float* b; half_t* wpk; int Cin, Cout, ks, Kpad;
};

extern "C" void kernel_launch(void* const* d_in, const int* in_sizes, int n_in,
                              void* d_out, int out_size, void* d_ws, size_t ws_size,
                              hipStream_t stream) {
    (void)in_sizes; (void)n_in; (void)out_size; (void)ws_size;

    size_t off = 0;
    auto carve = [&](size_t bytes) -> void* {
        void* p = (char*)d_ws + off;
        off += (bytes + 255) & ~(size_t)255;
        return p;
    };

    const size_t X_ELEMS   = (size_t)BATCH * 3 * 416 * 416;
    const size_t ACT_ELEMS = (size_t)BATCH * 64 * 208 * 208;   // max activation
    half_t* x16    = (half_t*)carve(X_ELEMS * 2);
    half_t* actA   = (half_t*)carve(ACT_ELEMS * 2);
    half_t* actB   = (half_t*)carve(ACT_ELEMS * 2);
    half_t* route0 = (half_t*)carve((size_t)BATCH * 256 * 52 * 52 * 2);
    half_t* route1 = (half_t*)carve((size_t)BATCH * 512 * 26 * 26 * 2);
    half_t* tip    = (half_t*)carve((size_t)BATCH * 256 * 52 * 52 * 2);
    half_t* raw    = (half_t*)carve((size_t)BATCH * 255 * 52 * 52 * 2);
    float*  det    = (float*) carve((size_t)BATCH * NTOT_ROWS * 6 * 4);
    float*  cand_s = (float*) carve((size_t)BATCH * CAND_MAX * 4);
    int*    cand_i = (int*)   carve((size_t)BATCH * CAND_MAX * 4);
    int*    cand_c = (int*)   carve(BATCH * 4);

    auto prep = [&](int wi, int si, int bi, int Cin, int Cout, int ks) -> ConvL {
        ConvL c;
        c.b = (const float*)d_in[bi];
        c.Cin = Cin; c.Cout = Cout; c.ks = ks;
        int Ktrue = Cin * ks * ks;
        c.Kpad = (Ktrue + 31) & ~31;
        c.wpk = (half_t*)carve((size_t)Cout * c.Kpad * 2);
        int tot = Cout * c.Kpad;
        pack_weights<<<(tot + 255) / 256, 256, 0, stream>>>(
            (const float*)d_in[wi], si >= 0 ? (const float*)d_in[si] : nullptr,
            c.wpk, Cout, Ktrue, c.Kpad);
        return c;
    };

    auto runconv = [&](const ConvL& c, const half_t* in, half_t* out,
                       int H, int stride, int leaky) -> int {
        int pad = c.ks / 2;
        int Ho = (H + 2 * pad - c.ks) / stride + 1;
        int Ntot = BATCH * Ho * Ho;
        dim3 g((Ntot + 127) / 128, (c.Cout + 63) / 64);
        int Ktrue = c.Cin * c.ks * c.ks;
        if (c.ks == 1)
            conv_wmma<1, 1><<<g, 256, 0, stream>>>(in, c.wpk, c.b, out, BATCH, c.Cin, H, H,
                                                   c.Cout, Ho, Ho, Ktrue, c.Kpad, leaky);
        else if (stride == 1)
            conv_wmma<3, 1><<<g, 256, 0, stream>>>(in, c.wpk, c.b, out, BATCH, c.Cin, H, H,
                                                   c.Cout, Ho, Ho, Ktrue, c.Kpad, leaky);
        else
            conv_wmma<3, 2><<<g, 256, 0, stream>>>(in, c.wpk, c.b, out, BATCH, c.Cin, H, H,
                                                   c.Cout, Ho, Ho, Ktrue, c.Kpad, leaky);
        return Ho;
    };

    // --- pack all weights (deterministic every call) ---
    ConvL stage[5], blocks[3][6], outs[3], trans[2];
    const int st_ci[5] = {3, 64, 128, 256, 512};
    const int st_co[5] = {64, 128, 256, 512, 1024};
    for (int i = 0; i < 5; ++i)
        stage[i] = prep(60 + 3 * i + 2, 60 + 3 * i + 1, 60 + 3 * i, st_ci[i], st_co[i], 3);
    const int blk_cin[3] = {1024, 768, 384};
    const int blk_ch[3]  = {512, 256, 128};
    for (int i = 0; i < 3; ++i) {
        int ch = blk_ch[i];
        const int ci[6] = {blk_cin[i], ch, 2 * ch, ch, 2 * ch, ch};
        const int co[6] = {ch, 2 * ch, ch, 2 * ch, ch, 2 * ch};
        const int ks[6] = {1, 3, 1, 3, 1, 3};
        for (int j = 0; j < 6; ++j)
            blocks[i][j] = prep(18 * i + 3 * j + 2, 18 * i + 3 * j + 1, 18 * i + 3 * j,
                                ci[j], co[j], ks[j]);
    }
    const int out_cin[3] = {1024, 512, 256};
    for (int i = 0; i < 3; ++i)
        outs[i] = prep(55 + 2 * i, -1, 54 + 2 * i, out_cin[i], 255, 1);
    trans[0] = prep(75 + 2, 75 + 1, 75, 512, 256, 1);
    trans[1] = prep(78 + 2, 78 + 1, 78, 256, 128, 1);

    // --- input f32 -> f16 ---
    convert_f32_to_f16<<<(int)((X_ELEMS + 255) / 256), 256, 0, stream>>>(
        (const float*)d_in[81], x16, (long long)X_ELEMS);

    // --- backbone ---
    int H = 416;
    H = runconv(stage[0], x16,   actA,   H, 2, 1);   // 208, 64
    H = runconv(stage[1], actA,  actB,   H, 2, 1);   // 104, 128
    H = runconv(stage[2], actB,  route0, H, 2, 1);   // 52, 256
    H = runconv(stage[3], route0, route1, H, 2, 1);  // 26, 512
    H = runconv(stage[4], route1, actA,  H, 2, 1);   // 13, 1024

    static const float ANCH[3][6] = {
        {10.f, 13.f, 16.f, 30.f, 33.f, 23.f},
        {30.f, 61.f, 62.f, 45.f, 59.f, 119.f},
        {116.f, 90.f, 156.f, 198.f, 373.f, 326.f}};
    const float STR[3] = {32.f, 16.f, 8.f};
    const long long BASE[3] = {0, 40560, 202800};
    const int HS[3] = {13, 26, 52};

    auto rundecode = [&](int i) {
        int h = HS[i];
        const float* A = ANCH[2 - i];
        long long tot = (long long)BATCH * h * h * 240;
        decode_kernel<<<(int)((tot + 255) / 256), 256, 0, stream>>>(
            raw, det, BATCH, h, h, STR[i], A[0], A[1], A[2], A[3], A[4], A[5],
            BASE[i], NTOT_ROWS);
    };
    auto runupcat = [&](const half_t* hsm, const half_t* rt, half_t* o,
                        int C1, int C2, int Hr) {
        long long tot = (long long)BATCH * (C1 + C2) * Hr * Hr;
        upcat_kernel<<<(int)((tot + 255) / 256), 256, 0, stream>>>(hsm, rt, o, BATCH, C1, C2, Hr, Hr);
    };

    // --- detection block 0 @13 ---
    runconv(blocks[0][0], actA, actB, 13, 1, 1);
    runconv(blocks[0][1], actB, actA, 13, 1, 1);
    runconv(blocks[0][2], actA, actB, 13, 1, 1);
    runconv(blocks[0][3], actB, actA, 13, 1, 1);
    runconv(blocks[0][4], actA, actB, 13, 1, 1);    // h (512)
    runconv(blocks[0][5], actB, tip,  13, 1, 1);    // tip (1024)
    runconv(outs[0], tip, raw, 13, 1, 0);
    rundecode(0);
    runconv(trans[0], actB, actA, 13, 1, 1);        // 256 @13
    runupcat(actA, route1, actB, 256, 512, 26);     // 768 @26

    // --- detection block 1 @26 ---
    runconv(blocks[1][0], actB, actA, 26, 1, 1);
    runconv(blocks[1][1], actA, actB, 26, 1, 1);
    runconv(blocks[1][2], actB, actA, 26, 1, 1);
    runconv(blocks[1][3], actA, actB, 26, 1, 1);
    runconv(blocks[1][4], actB, actA, 26, 1, 1);    // h (256)
    runconv(blocks[1][5], actA, tip,  26, 1, 1);    // tip (512)
    runconv(outs[1], tip, raw, 26, 1, 0);
    rundecode(1);
    runconv(trans[1], actA, actB, 26, 1, 1);        // 128 @26
    runupcat(actB, route0, actA, 128, 256, 52);     // 384 @52

    // --- detection block 2 @52 ---
    runconv(blocks[2][0], actA, actB, 52, 1, 1);
    runconv(blocks[2][1], actB, actA, 52, 1, 1);
    runconv(blocks[2][2], actA, actB, 52, 1, 1);
    runconv(blocks[2][3], actB, actA, 52, 1, 1);
    runconv(blocks[2][4], actA, actB, 52, 1, 1);    // h (128)
    runconv(blocks[2][5], actB, tip,  52, 1, 1);    // tip (256)
    runconv(outs[2], tip, raw, 52, 1, 0);
    rundecode(2);

    // --- NMS ---
    init_cnt<<<1, 32, 0, stream>>>(cand_c);
    long long ctot = (long long)BATCH * NTOT_ROWS;
    cand_kernel<<<(int)((ctot + 255) / 256), 256, 0, stream>>>(det, NTOT_ROWS, cand_s, cand_i, cand_c);
    nms_kernel<<<BATCH, 256, 0, stream>>>(det, NTOT_ROWS, cand_s, cand_i, cand_c, (float*)d_out);
}
